// Seq2SeqTransformer_14602888806720
// MI455X (gfx1250) — matrix operand order repaired and checked by reading, not compiled, not run on hardware
//
#include <hip/hip_runtime.h>
#include <hip/hip_bf16.h>
#include <math.h>

typedef __attribute__((ext_vector_type(16))) _Float16 v16h;
typedef __attribute__((ext_vector_type(8)))  float    v8f;
typedef __attribute__((ext_vector_type(4)))  _Float16 v4h;
typedef __attribute__((ext_vector_type(4)))  int      v4i;

#define D_MODEL 1024
#define N_HEAD  16
#define DH      64
#define FF_DIM  4096
#define NLAYER  6
#define BATCH   4
#define SEQ     512
#define NTOK    (BATCH*SEQ)
#define VOCAB   32000

#if defined(__HIP_DEVICE_COMPILE__) && \
    __has_builtin(__builtin_amdgcn_global_load_async_to_lds_b128) && \
    __has_builtin(__builtin_amdgcn_s_wait_asynccnt)
#define HAVE_ASYNC_LDS 1
#else
#define HAVE_ASYNC_LDS 0
#endif

#if HAVE_ASYNC_LDS
// Builtin prototype (from toolchain diagnostics):
//   (v4i addrspace(1)* src, v4i addrspace(3)* dst, imm offset, imm cpol)
typedef __attribute__((address_space(1))) v4i g_v4i;
typedef __attribute__((address_space(3))) v4i l_v4i;
// Flat-aperture identities: global addr == flat addr; LDS offset == flat[31:0].
__device__ static inline g_v4i* to_glob(const void* p) {
  return (g_v4i*)(unsigned long long)p;
}
__device__ static inline l_v4i* to_lds(void* p) {
  return (l_v4i*)(unsigned int)(unsigned long long)p;
}
#endif

// ---------------------------------------------------------------------------
// WMMA GEMM:  C[M,N] = A[M,K] @ Bt[N,K]^T + bias[N]   (optional ReLU)
// A, Bt are f16, both K-contiguous. Block tile 128x128, K-step 64,
// 256 threads = 8 waves (2x4); 16 v_wmma_f32_16x16x32_f16 per K-step.
// LDS tiles are double-buffered and filled with async global->LDS DMA
// (ASYNCcnt) when available; fragment reads are pairs of ds_load_b128.
// ---------------------------------------------------------------------------
__global__ __launch_bounds__(256) void gemm_kernel(
    const _Float16* __restrict__ A, const _Float16* __restrict__ Bt,
    const float* __restrict__ bias, float* __restrict__ C,
    int M, int N, int K, int relu)
{
  __shared__ alignas(16) _Float16 As [2][128][72];  // [m][k], row stride 144B
  __shared__ alignas(16) _Float16 BsT[2][128][72];  // [n][k]

  const int tid  = threadIdx.x;
  const int lane = tid & 31;
  const int wave = tid >> 5;
  const int wm   = (wave >> 2) * 64;   // wave M offset: 0/64
  const int wn   = (wave & 3) * 32;    // wave N offset: 0..96
  const int m0   = blockIdx.y * 128;
  const int n0   = blockIdx.x * 128;

  const int mrow = lane & 15;          // A row / B col / C col selector
  const int hsel = (lane & 16) >> 1;   // 0 or 8 (C m-half)
  const int kbB  = (lane & 16);        // 0 or 16 (B k-half)

  v8f acc[4][2];
  const v8f vzero = {0.f,0.f,0.f,0.f,0.f,0.f,0.f,0.f};
#pragma unroll
  for (int mt = 0; mt < 4; ++mt)
#pragma unroll
    for (int nt = 0; nt < 2; ++nt) acc[mt][nt] = vzero;

  auto compute = [&](int buf) {
    // B fragments: 16 contiguous halves per lane -> 2x ds_load_b128
    v16h bf[2][2];
#pragma unroll
    for (int nt = 0; nt < 2; ++nt) {
      int col = wn + nt * 16 + mrow;
#pragma unroll
      for (int kk = 0; kk < 2; ++kk)
#pragma unroll
        for (int v = 0; v < 8; ++v) {
          int k = kk * 32 + kbB + (v << 1);
          bf[nt][kk][2 * v]     = BsT[buf][col][k];
          bf[nt][kk][2 * v + 1] = BsT[buf][col][k + 1];
        }
    }
#pragma unroll
    for (int mt = 0; mt < 4; ++mt) {
      int row = wm + mt * 16 + mrow;
      v16h af[2];
#pragma unroll
      for (int kk = 0; kk < 2; ++kk)
#pragma unroll
        for (int v = 0; v < 8; ++v) {
          int k = kk * 32 + kbB + (v << 1);
          af[kk][2 * v]     = As[buf][row][k];
          af[kk][2 * v + 1] = As[buf][row][k + 1];
        }
      acc[mt][0] = __builtin_amdgcn_wmma_f32_16x16x32_f16(
          false, af[0], false, bf[0][0], (short)0, acc[mt][0], false, false);
      acc[mt][0] = __builtin_amdgcn_wmma_f32_16x16x32_f16(
          false, af[1], false, bf[0][1], (short)0, acc[mt][0], false, false);
      acc[mt][1] = __builtin_amdgcn_wmma_f32_16x16x32_f16(
          false, af[0], false, bf[1][0], (short)0, acc[mt][1], false, false);
      acc[mt][1] = __builtin_amdgcn_wmma_f32_16x16x32_f16(
          false, af[1], false, bf[1][1], (short)0, acc[mt][1], false, false);
    }
  };

#if HAVE_ASYNC_LDS
  // 8 async b128 DMA ops per wave per tile: 4 for A (128 rows x 128B),
  // 4 for B (128 cols x 128B). ASYNCcnt retires in order.
  auto issue = [&](int kt, int buf) {
#pragma unroll
    for (int i = 0; i < 4; ++i) {
      int lin = i * 256 + tid;
      int row = lin >> 3;
      int seg = (lin & 7) << 3;          // 8-half (16B) segment
      __builtin_amdgcn_global_load_async_to_lds_b128(
          to_glob(A + (size_t)(m0 + row) * K + kt + seg),
          to_lds(&As[buf][row][seg]), 0, 0);
    }
#pragma unroll
    for (int i = 0; i < 4; ++i) {
      int lin = i * 256 + tid;
      int row = lin >> 3;
      int seg = (lin & 7) << 3;
      __builtin_amdgcn_global_load_async_to_lds_b128(
          to_glob(Bt + (size_t)(n0 + row) * K + kt + seg),
          to_lds(&BsT[buf][row][seg]), 0, 0);
    }
  };

  const int nk = K >> 6;
  issue(0, 0);
  for (int t = 0; t < nk; ++t) {
    const int cb = t & 1;
    if (t + 1 < nk) {
      issue((t + 1) << 6, cb ^ 1);
      __builtin_amdgcn_s_wait_asynccnt(8);   // tile t's 8 DMAs complete
    } else {
      __builtin_amdgcn_s_wait_asynccnt(0);
    }
    __syncthreads();                          // tile t visible to all waves
    compute(cb);
    __syncthreads();                          // LDS reads done before overwrite
  }
#else
  // Fallback: registers -> LDS copies (f16 b128), single buffer.
  for (int kt = 0; kt < K; kt += 64) {
    uint4 ra[4], rb[4];
#pragma unroll
    for (int i = 0; i < 4; ++i) {
      int lin = i * 256 + tid;
      int row = lin >> 3;
      int seg = (lin & 7) << 3;
      ra[i] = *reinterpret_cast<const uint4*>(A + (size_t)(m0 + row) * K + kt + seg);
      rb[i] = *reinterpret_cast<const uint4*>(Bt + (size_t)(n0 + row) * K + kt + seg);
    }
    __syncthreads();
#pragma unroll
    for (int i = 0; i < 4; ++i) {
      int lin = i * 256 + tid;
      int row = lin >> 3;
      int seg = (lin & 7) << 3;
      *reinterpret_cast<uint4*>(&As [0][row][seg]) = ra[i];
      *reinterpret_cast<uint4*>(&BsT[0][row][seg]) = rb[i];
    }
    __syncthreads();
    compute(0);
  }
#endif

  // epilogue: C/D layout -> VGPR r holds M=r (lanes 0-15) / M=r+8 (lanes 16-31)
#pragma unroll
  for (int mt = 0; mt < 4; ++mt)
#pragma unroll
    for (int nt = 0; nt < 2; ++nt) {
      int col = n0 + wn + nt * 16 + mrow;
      float bv = bias ? bias[col] : 0.f;
#pragma unroll
      for (int r = 0; r < 8; ++r) {
        int row = m0 + wm + mt * 16 + r + hsel;
        float cv = acc[mt][nt][r] + bv;
        if (relu) cv = fmaxf(cv, 0.f);
        C[(size_t)row * N + col] = cv;
      }
    }
}

// ---------------------------------------------------------------------------
// f32 -> f16 elementwise (4 per thread)
// ---------------------------------------------------------------------------
__global__ __launch_bounds__(256) void cvt_f16_kernel(
    const float* __restrict__ in, _Float16* __restrict__ out)
{
  size_t i = ((size_t)blockIdx.x * 256 + threadIdx.x) * 4;
  float4 v = *reinterpret_cast<const float4*>(in + i);
  v4h h = {(_Float16)v.x, (_Float16)v.y, (_Float16)v.z, (_Float16)v.w};
  *reinterpret_cast<v4h*>(out + i) = h;
}

// ---------------------------------------------------------------------------
// B[K,N] f32 -> Bt[N,K] f16  (32x32 LDS tile transpose)
// ---------------------------------------------------------------------------
__global__ __launch_bounds__(256) void transpose_cvt_kernel(
    const float* __restrict__ in, _Float16* __restrict__ out, int K, int N)
{
  __shared__ _Float16 t[32][40];
  const int n0 = blockIdx.x * 32;
  const int k0 = blockIdx.y * 32;
  const int tx = threadIdx.x & 31;
  const int ty = threadIdx.x >> 5;     // 0..7
#pragma unroll
  for (int i = 0; i < 4; ++i)
    t[ty + i * 8][tx] = (_Float16)in[(size_t)(k0 + ty + i * 8) * N + n0 + tx];
  __syncthreads();
#pragma unroll
  for (int i = 0; i < 4; ++i)
    out[(size_t)(n0 + ty + i * 8) * K + k0 + tx] = t[tx][ty + i * 8];
}

// ---------------------------------------------------------------------------
// Embedding + sinusoidal positional encoding
// ---------------------------------------------------------------------------
__global__ __launch_bounds__(256) void embed_kernel(
    const int* __restrict__ tok, const float* __restrict__ emb,
    float* __restrict__ out)
{
  int idx = blockIdx.x * 256 + threadIdx.x;   // over NTOK * D_MODEL
  int d = idx & (D_MODEL - 1);
  int t = idx >> 10;
  int pos = t & (SEQ - 1);
  int token = tok[t];
  float i2 = (float)(d & ~1);
  float ang = (float)pos * __expf(-(i2 * (1.0f / (float)D_MODEL)) * 9.210340371976184f);
  float pe = (d & 1) ? __cosf(ang) : __sinf(ang);
  out[idx] = emb[(size_t)token * D_MODEL + d] + pe;
}

// ---------------------------------------------------------------------------
// Softmax attention: one block per (b, h, q). Q/K/V are [B, L, H*DH].
// ---------------------------------------------------------------------------
__global__ __launch_bounds__(256) void attn_kernel(
    const float* __restrict__ Q, const float* __restrict__ Km,
    const float* __restrict__ V, float* __restrict__ O,
    const unsigned char* __restrict__ pad, int causal)
{
  __shared__ float qv[DH];
  __shared__ float sc[SEQ];
  __shared__ float red[256];

  const int qi = blockIdx.x & (SEQ - 1);
  const int h  = (blockIdx.x >> 9) & (N_HEAD - 1);
  const int b  = blockIdx.x >> 13;
  const int tid = threadIdx.x;
  const size_t qoff = (size_t)(b * SEQ + qi) * D_MODEL + h * DH;

  if (tid < DH) qv[tid] = Q[qoff + tid];
  __syncthreads();

  for (int k = tid; k < SEQ; k += 256) {
    const float* kr = Km + (size_t)(b * SEQ + k) * D_MODEL + h * DH;
    float s = 0.f;
#pragma unroll 8
    for (int d = 0; d < DH; ++d) s += qv[d] * kr[d];
    s *= 0.125f;  // 1/sqrt(64)
    if (causal && k > qi) s = -__builtin_inff();
    if (pad[b * SEQ + k]) s = -__builtin_inff();
    sc[k] = s;
  }
  __syncthreads();

  float m = -__builtin_inff();
  for (int k = tid; k < SEQ; k += 256) m = fmaxf(m, sc[k]);
  red[tid] = m; __syncthreads();
  for (int o = 128; o > 0; o >>= 1) {
    if (tid < o) red[tid] = fmaxf(red[tid], red[tid + o]);
    __syncthreads();
  }
  m = red[0]; __syncthreads();

  float s = 0.f;
  for (int k = tid; k < SEQ; k += 256) {
    float e = __expf(sc[k] - m);
    sc[k] = e; s += e;
  }
  red[tid] = s; __syncthreads();
  for (int o = 128; o > 0; o >>= 1) {
    if (tid < o) red[tid] += red[tid + o];
    __syncthreads();
  }
  const float rsum = 1.0f / red[0];
  __syncthreads();

  const int d = tid & (DH - 1);
  const int c = tid >> 6;
  float p = 0.f;
  for (int k = c * (SEQ / 4); k < (c + 1) * (SEQ / 4); ++k)
    p += sc[k] * V[(size_t)(b * SEQ + k) * D_MODEL + h * DH + d];
  red[tid] = p; __syncthreads();
  if (c == 0)
    O[qoff + d] = (red[d] + red[d + 64] + red[d + 128] + red[d + 192]) * rsum;
}

// ---------------------------------------------------------------------------
// out = LayerNorm(x + dx) * g + b   (one block per token row, D=1024)
// ---------------------------------------------------------------------------
__global__ __launch_bounds__(256) void add_ln_kernel(
    const float* __restrict__ x, const float* __restrict__ dx,
    const float* __restrict__ g, const float* __restrict__ b,
    float* __restrict__ out)
{
  __shared__ float red[256];
  const int row = blockIdx.x;
  const int tid = threadIdx.x;
  const float* xr = x  + (size_t)row * D_MODEL;
  const float* dr = dx + (size_t)row * D_MODEL;

  float v[4];
  float s = 0.f;
#pragma unroll
  for (int i = 0; i < 4; ++i) {
    int c = tid + i * 256;
    v[i] = xr[c] + dr[c];
    s += v[i];
  }
  red[tid] = s; __syncthreads();
  for (int o = 128; o > 0; o >>= 1) {
    if (tid < o) red[tid] += red[tid + o];
    __syncthreads();
  }
  const float mean = red[0] * (1.0f / D_MODEL);
  __syncthreads();

  s = 0.f;
#pragma unroll
  for (int i = 0; i < 4; ++i) { float dvi = v[i] - mean; s += dvi * dvi; }
  red[tid] = s; __syncthreads();
  for (int o = 128; o > 0; o >>= 1) {
    if (tid < o) red[tid] += red[tid + o];
    __syncthreads();
  }
  const float rstd = rsqrtf(red[0] * (1.0f / D_MODEL) + 1e-5f);

#pragma unroll
  for (int i = 0; i < 4; ++i) {
    int c = tid + i * 256;
    out[(size_t)row * D_MODEL + c] = (v[i] - mean) * rstd * g[c] + b[c];
  }
}

// ---------------------------------------------------------------------------
// Host orchestration
// ---------------------------------------------------------------------------
extern "C" void kernel_launch(void* const* d_in, const int* in_sizes, int n_in,
                              void* d_out, int out_size, void* d_ws, size_t ws_size,
                              hipStream_t stream)
{
  (void)in_sizes; (void)n_in; (void)out_size; (void)ws_size;
  auto P = [&](int i) { return (const float*)d_in[i]; };

  // pytree (alphabetical dict) flattening order -- see previous rounds.
  const int* srcT = (const int*)d_in[46];
  const unsigned char* srcPad = (const unsigned char*)d_in[47];
  const int* tgtT = (const int*)d_in[48];
  const unsigned char* tgtPad = (const unsigned char*)d_in[49];

  const size_t DD = (size_t)D_MODEL * D_MODEL;
  const size_t DV = D_MODEL;
  const size_t DF = (size_t)D_MODEL * FF_DIM;
  const size_t FV = FF_DIM;

  float* ws = (float*)d_ws;
  const size_t SZ = (size_t)NTOK * D_MODEL;
  float* X   = ws;            // encoder activations -> memory
  float* X2  = X  + SZ;       // ping buffer
  float* Y2  = X2 + SZ;       // pong buffer
  float* Qb  = Y2 + SZ;
  float* Kb  = Qb + SZ;
  float* Vb  = Kb + SZ;
  float* Ab  = Vb + SZ;       // attention context output
  float* Tb  = Ab + SZ;       // projection output
  float* FFb = Tb + SZ;       // NTOK x FF_DIM
  _Float16* Ah  = (_Float16*)(FFb + (size_t)NTOK * FF_DIM);  // f16 A scratch
  _Float16* Bth = Ah + (size_t)NTOK * FF_DIM;                // f16 B^T scratch

  dim3 blk(256);
  auto cvtA = [&](const float* Af, int M, int K) {
    cvt_f16_kernel<<<(M * K) / 1024, blk, 0, stream>>>(Af, Ah);
  };
  auto trB = [&](const float* Bf, int K, int N) {
    transpose_cvt_kernel<<<dim3(N / 32, K / 32), blk, 0, stream>>>(Bf, Bth, K, N);
  };
  auto mm = [&](const float* bias, float* C, int M, int N, int K, int relu) {
    gemm_kernel<<<dim3(N / 128, M / 128), blk, 0, stream>>>(Ah, Bth, bias, C, M, N, K, relu);
  };

  const int embBlocks  = (NTOK * D_MODEL) / 256;
  const int attnBlocks = BATCH * N_HEAD * SEQ;

  // ---- encoder ----
  embed_kernel<<<embBlocks, blk, 0, stream>>>(srcT, P(44), X);
  for (int l = 0; l < NLAYER; ++l) {
    cvtA(X, NTOK, D_MODEL);
    trB(P(37) + l * DD, D_MODEL, D_MODEL); mm(P(36) + l * DV, Qb, NTOK, D_MODEL, D_MODEL, 0);
    trB(P(33) + l * DD, D_MODEL, D_MODEL); mm(P(32) + l * DV, Kb, NTOK, D_MODEL, D_MODEL, 0);
    trB(P(39) + l * DD, D_MODEL, D_MODEL); mm(P(38) + l * DV, Vb, NTOK, D_MODEL, D_MODEL, 0);
    attn_kernel<<<attnBlocks, blk, 0, stream>>>(Qb, Kb, Vb, Ab, srcPad, 0);
    cvtA(Ab, NTOK, D_MODEL);
    trB(P(35) + l * DD, D_MODEL, D_MODEL); mm(P(34) + l * DV, Tb, NTOK, D_MODEL, D_MODEL, 0);
    add_ln_kernel<<<NTOK, blk, 0, stream>>>(X, Tb, P(29) + l * DV, P(28) + l * DV, X2);
    cvtA(X2, NTOK, D_MODEL);
    trB(P(40) + l * DF, D_MODEL, FF_DIM); mm(P(26) + l * FV, FFb, NTOK, FF_DIM, D_MODEL, 1);
    cvtA(FFb, NTOK, FF_DIM);
    trB(P(41) + l * DF, FF_DIM, D_MODEL); mm(P(27) + l * DV, Tb, NTOK, D_MODEL, FF_DIM, 0);
    add_ln_kernel<<<NTOK, blk, 0, stream>>>(X2, Tb, P(31) + l * DV, P(30) + l * DV, X);
  }
  // memory = X

  // ---- decoder ----
  float* cur = X2;
  float* oth = Y2;
  embed_kernel<<<embBlocks, blk, 0, stream>>>(tgtT, P(45), cur);
  for (int l = 0; l < NLAYER; ++l) {
    // masked self-attention
    cvtA(cur, NTOK, D_MODEL);
    trB(P(21) + l * DD, D_MODEL, D_MODEL); mm(P(20) + l * DV, Qb, NTOK, D_MODEL, D_MODEL, 0);
    trB(P(17) + l * DD, D_MODEL, D_MODEL); mm(P(16) + l * DV, Kb, NTOK, D_MODEL, D_MODEL, 0);
    trB(P(23) + l * DD, D_MODEL, D_MODEL); mm(P(22) + l * DV, Vb, NTOK, D_MODEL, D_MODEL, 0);
    attn_kernel<<<attnBlocks, blk, 0, stream>>>(Qb, Kb, Vb, Ab, tgtPad, 1);
    cvtA(Ab, NTOK, D_MODEL);
    trB(P(19) + l * DD, D_MODEL, D_MODEL); mm(P(18) + l * DV, Tb, NTOK, D_MODEL, D_MODEL, 0);
    add_ln_kernel<<<NTOK, blk, 0, stream>>>(cur, Tb, P(11) + l * DV, P(10) + l * DV, oth);
    // cross-attention (q from oth, k/v from encoder memory X)
    cvtA(oth, NTOK, D_MODEL);
    trB(P(7) + l * DD, D_MODEL, D_MODEL); mm(P(6) + l * DV, Qb, NTOK, D_MODEL, D_MODEL, 0);
    cvtA(X, NTOK, D_MODEL);
    trB(P(3) + l * DD, D_MODEL, D_MODEL); mm(P(2) + l * DV, Kb, NTOK, D_MODEL, D_MODEL, 0);
    trB(P(9) + l * DD, D_MODEL, D_MODEL); mm(P(8) + l * DV, Vb, NTOK, D_MODEL, D_MODEL, 0);
    attn_kernel<<<attnBlocks, blk, 0, stream>>>(Qb, Kb, Vb, Ab, srcPad, 0);
    cvtA(Ab, NTOK, D_MODEL);
    trB(P(5) + l * DD, D_MODEL, D_MODEL); mm(P(4) + l * DV, Tb, NTOK, D_MODEL, D_MODEL, 0);
    add_ln_kernel<<<NTOK, blk, 0, stream>>>(oth, Tb, P(13) + l * DV, P(12) + l * DV, cur);
    // FFN
    cvtA(cur, NTOK, D_MODEL);
    trB(P(24) + l * DF, D_MODEL, FF_DIM); mm(P(0) + l * FV, FFb, NTOK, FF_DIM, D_MODEL, 1);
    cvtA(FFb, NTOK, FF_DIM);
    trB(P(25) + l * DF, FF_DIM, D_MODEL); mm(P(1) + l * DV, Tb, NTOK, D_MODEL, FF_DIM, 0);
    add_ln_kernel<<<NTOK, blk, 0, stream>>>(cur, Tb, P(15) + l * DV, P(14) + l * DV, oth);
    float* t = cur; cur = oth; oth = t;
  }

  // ---- output projection: [2048,1024] @ [1024,32000] + out_b ----
  cvtA(cur, NTOK, D_MODEL);
  trB(P(43), D_MODEL, VOCAB);
  mm(P(42), (float*)d_out, NTOK, VOCAB, D_MODEL, 0);
}
